// UCBAttention_50190987821234
// MI455X (gfx1250) — matrix operand order repaired and checked by reading, not compiled, not run on hardware
//
#include <hip/hip_runtime.h>

// ---------------------------------------------------------------------------
// UCB attention, fused, bf16 WMMA path for gfx1250 (MI455X).
//
// Shapes: B=8, N=784, C=768, H=12, Dh=64, TOPK=100, counter=1000 -> UCB branch.
// d_in:  [0] x f32 (B,N,C)          [1] ucb_count_score f32 (H,N,N)
//        [2] qkv_w f32 (3C,C)       [3] proj_w f32 (C,C)
//        [4] proj_b f32 (C)         [5] counter i32  [6] ucb_enabled i32
// d_out: out f32 (B,N,C) ++ score_delta f32 (H,N,N)
//
// GEMMs are 16x64-per-wave bf16 WMMA strips (A fragment reused across 4
// accumulators) -> 4 v_wmma per 10 global_load_b128.  Attention is fully
// fused per (b,h,16-query tile): scores live in LDS as bf16, top-k threshold
// is a 16-step in-register binary search per wave, PV is WMMA from LDS probs
// against a K-contiguous V^T.
// ---------------------------------------------------------------------------

typedef unsigned short u16;
typedef __attribute__((ext_vector_type(8)))  u16    ushort8;
typedef __attribute__((ext_vector_type(16))) __bf16 v16bf;
typedef __attribute__((ext_vector_type(8)))  float  v8f;

#define BB   8
#define NN   784
#define CC   768
#define HH   12
#define DH   64
#define C3   2304
#define BN   (BB * NN)     // 6272
#define NKP  800           // key dim padded to 25*32 for the PV K-loop
#define TOPK 100

// ---- bf16 helpers ---------------------------------------------------------
static __device__ __forceinline__ u16 f32_to_bf16(float f) {
  union { float f; unsigned u; } v; v.f = f;
  unsigned u = v.u;
  return (u16)((u + 0x7FFFu + ((u >> 16) & 1u)) >> 16);   // RNE
}
static __device__ __forceinline__ float bf16_to_f32(u16 h) {
  union { unsigned u; float f; } v; v.u = ((unsigned)h) << 16;
  return v.f;
}
// monotonic map bf16 bits -> unsigned (for descending top-k compares)
static __device__ __forceinline__ unsigned sort16(u16 h) {
  unsigned u = h;
  return (u & 0x8000u) ? (u ^ 0xFFFFu) : (u | 0x8000u);
}

// ---- WMMA fragments -------------------------------------------------------
union FragBF { v16bf bf; ushort8 h[2]; };

// A: 16xK row-major tile, per CDNA5 16-bit A layout (lane m=lane&15,
// halves e: K = (e/8)*16 + (lane>=16 ? 8 : 0) + e%8) -> two b128 loads.
static __device__ __forceinline__ FragBF load_a(const u16* __restrict__ A,
                                                int lda, int m0, int k0, int lane) {
  FragBF f;
  const u16* p = A + (size_t)(m0 + (lane & 15)) * lda + k0 + (lane >> 4) * 8;
  f.h[0] = *(const ushort8*)p;
  f.h[1] = *(const ushort8*)(p + 16);
  return f;
}
// B: Kx16 tile supplied as Bt (N x K row-major, contiguous in K).
// Layout: col n=lane&15, halves e: K = (e/8)*16 + (lane>=16 ? 8 : 0) + e%8 -> two b128 loads.
static __device__ __forceinline__ FragBF load_bt(const u16* __restrict__ Bt,
                                                 int ldb, int n0, int k0, int lane) {
  FragBF f;
  const u16* p = Bt + (size_t)(n0 + (lane & 15)) * ldb + k0 + (lane >> 4) * 8;
  f.h[0] = *(const ushort8*)p;
  f.h[1] = *(const ushort8*)(p + 16);
  return f;
}
static __device__ __forceinline__ v8f wmma_bf16(FragBF a, FragBF b, v8f c) {
  return __builtin_amdgcn_wmma_f32_16x16x32_bf16(false, a.bf, false, b.bf,
                                                 (short)0, c, false, false);
}

// ---------------------------------------------------------------------------
// Kernel 0: fp32 -> bf16 conversion of x, qkv_w, proj_w
// ---------------------------------------------------------------------------
__global__ void ucb_prep_kernel(const float* __restrict__ x,
                                const float* __restrict__ qkvw,
                                const float* __restrict__ projw,
                                u16* __restrict__ xb, u16* __restrict__ qwb,
                                u16* __restrict__ pwb) {
  const int n1 = BN * CC, n2 = C3 * CC, n3 = CC * CC;
  for (int i = blockIdx.x * blockDim.x + threadIdx.x; i < n1 + n2 + n3;
       i += gridDim.x * blockDim.x) {
    if (i < n1)            xb[i]            = f32_to_bf16(x[i]);
    else if (i < n1 + n2)  qwb[i - n1]      = f32_to_bf16(qkvw[i - n1]);
    else                   pwb[i - n1 - n2] = f32_to_bf16(projw[i - n1 - n2]);
  }
}

// ---------------------------------------------------------------------------
// Kernel 1: zero score_delta (atomics accumulate into it)
// ---------------------------------------------------------------------------
__global__ void ucb_zero_kernel(float* __restrict__ p, int n) {
  for (int i = blockIdx.x * blockDim.x + threadIdx.x; i < n;
       i += gridDim.x * blockDim.x) p[i] = 0.0f;
}

// ---------------------------------------------------------------------------
// Kernel 2: qkv projection GEMM. (BN x C) @ (C3 x C)^T, bf16 WMMA, f32 acc.
// 16x64 strip per wave: one A fragment feeds 4 accumulators.
// Writes q,k row-major (BH,N,Dh) and V transposed (BH,Dh,NKP) so the PV
// B-operand is contiguous along K(=keys).
// Strips: 392 * (2304/64) = 392*36 = 14112; 8 waves / block -> 1764 blocks.
// ---------------------------------------------------------------------------
__global__ __launch_bounds__(256) void ucb_gemm_qkv_kernel(
    const u16* __restrict__ A, const u16* __restrict__ W,
    u16* __restrict__ qb, u16* __restrict__ kb, u16* __restrict__ vtb) {
  const int lane = threadIdx.x & 31, wid = threadIdx.x >> 5;
  const int strip = blockIdx.x * 8 + wid;
  const int GN = C3 / 64;                        // 36
  const int mt = strip / GN, n0 = (strip % GN) * 64;

  v8f acc[4];
#pragma unroll
  for (int t = 0; t < 4; ++t) acc[t] = (v8f){0.f,0.f,0.f,0.f,0.f,0.f,0.f,0.f};

  for (int k0 = 0; k0 < CC; k0 += 32) {
    FragBF a = load_a(A, CC, mt * 16, k0, lane);
#pragma unroll
    for (int t = 0; t < 4; ++t) {
      FragBF b = load_bt(W, CC, n0 + t * 16, k0, lane);
      acc[t] = wmma_bf16(a, b, acc[t]);
    }
  }

  const int n  = lane & 15;
  const int mb = (lane >> 4) * 8;
#pragma unroll
  for (int t = 0; t < 4; ++t) {
    const int j = n0 + t * 16 + n;
    const int sec = j / CC, jj = j % CC, h = jj >> 6, dh = jj & 63;
#pragma unroll
    for (int v = 0; v < 8; ++v) {
      const int rowg = mt * 16 + mb + v;
      const int b = rowg / NN, nq = rowg % NN;
      const u16 val = f32_to_bf16(acc[t][v]);
      const size_t bh = (size_t)(b * HH + h);
      if (sec == 0)      qb[(bh * NN + nq) * DH + dh] = val;
      else if (sec == 1) kb[(bh * NN + nq) * DH + dh] = val;
      else               vtb[(bh * DH + dh) * NKP + nq] = val;
    }
  }
}

// ---------------------------------------------------------------------------
// Kernel 3: fused attention.  One block = one (b,h, 16-query tile).
// 16 waves: phase A each wave computes score key-tiles via WMMA + adds UCB;
// phase B/C one wave per query row does 16-bit binary-search top-k threshold,
// mask+normalize in place, atomic score_delta; phase D waves 0-3 do PV WMMA.
// ---------------------------------------------------------------------------
__global__ __launch_bounds__(512) void ucb_attn_kernel(
    const u16* __restrict__ qb, const u16* __restrict__ kb,
    const u16* __restrict__ vtb, const float* __restrict__ ucb,
    const int* __restrict__ counter, float* __restrict__ sdelta,
    u16* __restrict__ ctx) {
  __shared__ u16 s_sc[16][NKP];    // scores (bf16), later pruned probs in place
  __shared__ u16 s_key[16][NKP];   // sortable(ucb_scores) 16-bit keys

  const int lane = threadIdx.x & 31, wid = threadIdx.x >> 5;
  const int qt = blockIdx.x % (NN / 16);
  const int bh = blockIdx.x / (NN / 16);
  const int h = bh % HH, b = bh / HH;
  const int q0 = qt * 16;

  const u16* qm = qb + (size_t)bh * NN * DH;
  const u16* km = kb + (size_t)bh * NN * DH;
  const u16* vt = vtb + (size_t)bh * DH * NKP;
  const float logC = logf((float)counter[0]);   // counter>500 -> UCB branch

  // ---- Phase A: scores = (Q K^T) * scale, keys = sortable(scores + ucb)
  FragBF a0 = load_a(qm, DH, q0, 0, lane);
  FragBF a1 = load_a(qm, DH, q0, 32, lane);
  for (int t = wid; t < 50; t += 16) {
    if (t == 49) {                       // zero the 784..799 padding columns
      for (int r = lane; r < 16 * 16; r += 32) {
        s_sc[r >> 4][784 + (r & 15)] = 0;
        s_key[r >> 4][784 + (r & 15)] = 0;
      }
      continue;
    }
    if (t + 16 < 49)                     // gfx1250 global_prefetch of next K tile
      __builtin_prefetch(km + (size_t)(t + 16) * 16 * DH + (lane << 5), 0, 1);
    v8f acc = {0.f, 0.f, 0.f, 0.f, 0.f, 0.f, 0.f, 0.f};
    FragBF b0 = load_bt(km, DH, t * 16, 0, lane);
    FragBF b1 = load_bt(km, DH, t * 16, 32, lane);
    acc = wmma_bf16(a0, b0, acc);
    acc = wmma_bf16(a1, b1, acc);
    const int key = t * 16 + (lane & 15);
    const int mb = (lane >> 4) * 8;
#pragma unroll
    for (int v = 0; v < 8; ++v) {
      const int row = mb + v;
      const float s = acc[v] * 0.125f;                    // Dh^-0.5
      const float cnt = ucb[((size_t)h * NN + (q0 + row)) * NN + key];
      const float ub = sqrtf(logC / (cnt + 1e-6f));       // BETA = 1
      s_sc[row][key] = f32_to_bf16(s);
      s_key[row][key] = (u16)sort16(f32_to_bf16(s + ub));
    }
  }
  __syncthreads();

  // ---- Phase B: per-row top-k threshold (wave `wid` owns row `wid`)
  const int r = wid;
  unsigned short kv[25];
#pragma unroll
  for (int i = 0; i < 25; ++i) kv[i] = s_key[r][lane + i * 32];

  unsigned thr = 0;
  for (int bit = 15; bit >= 0; --bit) {
    const unsigned cand = thr | (1u << bit);
    int c = 0;
#pragma unroll
    for (int i = 0; i < 25; ++i) c += ((unsigned)kv[i] >= cand);
    for (int off = 16; off; off >>= 1) c += __shfl_xor(c, off, 32);
    if (c >= TOPK) thr = cand;
  }

  // ---- Phase C: mask, normalize in place, accumulate score_delta
  float sum = 0.f;
#pragma unroll
  for (int i = 0; i < 25; ++i) {
    const int j = lane + i * 32;
    if (j < NN && (unsigned)kv[i] >= thr) sum += bf16_to_f32(s_sc[r][j]);
  }
  for (int off = 16; off; off >>= 1) sum += __shfl_xor(sum, off, 32);
  const float inv = 1.f / (sum + 1e-8f);
#pragma unroll
  for (int i = 0; i < 25; ++i) {
    const int j = lane + i * 32;
    const bool sel = (j < NN) && ((unsigned)kv[i] >= thr);
    const float p = sel ? bf16_to_f32(s_sc[r][j]) * inv : 0.f;
    s_sc[r][j] = f32_to_bf16(p);
    if (sel) atomicAdd(&sdelta[((size_t)h * NN + (q0 + r)) * NN + j], 1.0f);
  }
  __syncthreads();

  // ---- Phase D: context = P(16x800) @ V(800x64); waves 0..3 own Dh tiles
  if (wid < 4) {
    v8f acc = {0.f, 0.f, 0.f, 0.f, 0.f, 0.f, 0.f, 0.f};
    for (int kk = 0; kk < NKP; kk += 32) {
      FragBF a = load_a(&s_sc[0][0], NKP, 0, kk, lane);      // LDS
      FragBF bf = load_bt(vt, NKP, wid * 16, kk, lane);      // V^T, contig in K
      acc = wmma_bf16(a, bf, acc);
    }
    const int n = lane & 15, mb = (lane >> 4) * 8;
#pragma unroll
    for (int v = 0; v < 8; ++v) {
      const int row = mb + v;
      ctx[((size_t)b * NN + (q0 + row)) * CC + h * DH + wid * 16 + n] =
          f32_to_bf16(acc[v]);
    }
  }
}

// ---------------------------------------------------------------------------
// Kernel 4: out = ctx @ proj_w^T + proj_b, f32 output.
// 16x64 strip per wave.  Strips: 392 * (768/64) = 392*12 = 4704 -> 588 blocks.
// ---------------------------------------------------------------------------
__global__ __launch_bounds__(256) void ucb_gemm_out_kernel(
    const u16* __restrict__ A, const u16* __restrict__ W,
    const float* __restrict__ bias, float* __restrict__ out) {
  const int lane = threadIdx.x & 31, wid = threadIdx.x >> 5;
  const int strip = blockIdx.x * 8 + wid;
  const int GN = CC / 64;                       // 12
  const int mt = strip / GN, n0 = (strip % GN) * 64;

  v8f acc[4];
#pragma unroll
  for (int t = 0; t < 4; ++t) acc[t] = (v8f){0.f,0.f,0.f,0.f,0.f,0.f,0.f,0.f};

  for (int k0 = 0; k0 < CC; k0 += 32) {
    FragBF a = load_a(A, CC, mt * 16, k0, lane);
#pragma unroll
    for (int t = 0; t < 4; ++t) {
      FragBF b = load_bt(W, CC, n0 + t * 16, k0, lane);
      acc[t] = wmma_bf16(a, b, acc[t]);
    }
  }

  const int n = lane & 15, mb = (lane >> 4) * 8;
#pragma unroll
  for (int t = 0; t < 4; ++t) {
    const int j = n0 + t * 16 + n;
    const float bj = bias[j];
#pragma unroll
    for (int v = 0; v < 8; ++v) {
      const int rowg = mt * 16 + mb + v;
      out[(size_t)rowg * CC + j] = acc[t][v] + bj;
    }
  }
}

// ---------------------------------------------------------------------------
extern "C" void kernel_launch(void* const* d_in, const int* in_sizes, int n_in,
                              void* d_out, int out_size, void* d_ws,
                              size_t ws_size, hipStream_t stream) {
  (void)in_sizes; (void)n_in; (void)out_size; (void)ws_size;
  const float* x      = (const float*)d_in[0];
  const float* ucb    = (const float*)d_in[1];
  const float* qkvw   = (const float*)d_in[2];
  const float* projw  = (const float*)d_in[3];
  const float* projb  = (const float*)d_in[4];
  const int*   counter= (const int*)d_in[5];

  float* out    = (float*)d_out;                          // (B,N,C)
  float* sdelta = out + (size_t)BN * CC;                  // (H,N,N)

  // workspace layout (bf16 buffers)
  char* ws = (char*)d_ws;
  size_t off = 0;
  auto take = [&](size_t elems) { u16* p = (u16*)(ws + off);
                                  off += elems * sizeof(u16);
                                  off = (off + 255) & ~(size_t)255; return p; };
  u16* xb  = take((size_t)BN * CC);
  u16* qwb = take((size_t)C3 * CC);
  u16* pwb = take((size_t)CC * CC);
  u16* qb  = take((size_t)BB * HH * NN * DH);
  u16* kb  = take((size_t)BB * HH * NN * DH);
  u16* vtb = take((size_t)BB * HH * DH * NKP);
  u16* ctx = take((size_t)BN * CC);

  ucb_prep_kernel<<<2048, 256, 0, stream>>>(x, qkvw, projw, xb, qwb, pwb);
  ucb_zero_kernel<<<4096, 256, 0, stream>>>(sdelta, HH * NN * NN);
  ucb_gemm_qkv_kernel<<<(392 * 36) / 8, 256, 0, stream>>>(xb, qwb, qb, kb, vtb);
  ucb_attn_kernel<<<BB * HH * (NN / 16), 512, 0, stream>>>(qb, kb, vtb, ucb,
                                                           counter, sdelta, ctx);
  ucb_gemm_out_kernel<<<(392 * 12) / 8, 256, 0, stream>>>(ctx, pwb, projb, out);
}
